// Block_71932112273752
// MI455X (gfx1250) — matrix operand order repaired and checked by reading, not compiled, not run on hardware
//
#include <hip/hip_runtime.h>
#include <math.h>

typedef _Float16 f16_t;
typedef __attribute__((ext_vector_type(16))) _Float16 v16h;
typedef __attribute__((ext_vector_type(8)))  _Float16 v8h;
typedef __attribute__((ext_vector_type(8)))  float    v8f;

#define DIM      768
#define HEADS    12
#define HEAD_DIM 64
#define HIDDEN   3072
#define NTOK     577
#define BATCH    64
#define MROWS    (BATCH * NTOK)   // 36928
#define QKV_W    (3 * DIM)        // 2304

__device__ __forceinline__ v16h concat8(v8h lo, v8h hi) {
    return __builtin_shufflevector(lo, hi, 0,1,2,3,4,5,6,7,8,9,10,11,12,13,14,15);
}
__device__ __forceinline__ v8h ld8(const f16_t* p) {
    return *(const v8h*)p;
}

// ---------------------------------------------------------------- weight cvt
__global__ __launch_bounds__(256) void cvt_f32_f16(const float* __restrict__ s,
                                                   f16_t* __restrict__ d, int n) {
    int i = blockIdx.x * 256 + threadIdx.x;
    if (i < n) d[i] = (f16_t)s[i];
}

// ---------------------------------------------------------------- layernorm
// one wave32 per 768-element row; 24 elems/lane; shfl reductions; f16 out
__global__ __launch_bounds__(256) void layernorm_to_f16(
    const float* __restrict__ x, const float* __restrict__ g,
    const float* __restrict__ b, f16_t* __restrict__ out, int rows) {
    int wave = threadIdx.x >> 5, lane = threadIdx.x & 31;
    int row = blockIdx.x * 8 + wave;
    if (row >= rows) return;
    const float* xr = x + (size_t)row * DIM + lane * 24;
    float vals[24];
#pragma unroll
    for (int i = 0; i < 6; ++i) {
        float4 v = *(const float4*)(xr + i * 4);
        vals[i*4+0] = v.x; vals[i*4+1] = v.y; vals[i*4+2] = v.z; vals[i*4+3] = v.w;
    }
    float s = 0.f, ss = 0.f;
#pragma unroll
    for (int i = 0; i < 24; ++i) { s += vals[i]; ss += vals[i] * vals[i]; }
#pragma unroll
    for (int m = 16; m >= 1; m >>= 1) {
        s  += __shfl_xor(s,  m, 32);
        ss += __shfl_xor(ss, m, 32);
    }
    float mu  = s * (1.f / DIM);
    float var = ss * (1.f / DIM) - mu * mu;
    float rs  = rsqrtf(var + 1e-5f);
    f16_t* orow = out + (size_t)row * DIM + lane * 24;
#pragma unroll
    for (int i = 0; i < 24; ++i) {
        int c = lane * 24 + i;
        orow[i] = (f16_t)((vals[i] - mu) * rs * g[c] + b[c]);
    }
}

// ---------------------------------------------------------------- WMMA GEMM
// C[M,N] = A[M,K](f16) * B[K,N](f16), LDS-tiled 128x128x32, 8 waves,
// wave tile 32x64 (2x4 wmma tiles), software-pipelined global->LDS staging.
// EPI: 0=QKV(f16 out) 1=proj(+bias+resid, f32 out)
//      2=fc1(+bias+gelu, f16 out) 3=fc2(+bias+resid, f32 out)
#define TM 128
#define TN 128
#define TK 32
#define LDR 40   // 32 + 8 pad (80B row stride, 16B aligned)

template<int EPI>
__global__ __launch_bounds__(256) void gemm_f16_wmma(
    const f16_t* __restrict__ A, const f16_t* __restrict__ Bw,
    const float* __restrict__ bias, const float* __restrict__ resid,
    float* __restrict__ outF, f16_t* __restrict__ outB,
    int M, int N, int K) {
    __shared__ __attribute__((aligned(16))) f16_t As[TM][LDR];
    __shared__ __attribute__((aligned(16))) f16_t Bs[TN][LDR];  // transposed: [n][k]
    const int tid   = threadIdx.x;
    const int wave  = tid >> 5;
    const int lane  = tid & 31;
    const int bm    = blockIdx.y * TM;
    const int bn    = blockIdx.x * TN;
    const int wm    = (wave >> 1) * 32;
    const int wn    = (wave & 1) * 64;
    const int lrow  = lane & 15;
    const int lhalf = lane >> 4;

    // per-thread staging coordinates (2 units each for A and B)
    int arow[2], akc[2], bkk[2], bnc[2];
#pragma unroll
    for (int it = 0; it < 2; ++it) {
        int id = tid + it * 256;          // 0..511
        arow[it] = id >> 2;               // 0..127
        akc[it]  = (id & 3) * 8;          // 0/8/16/24
        bkk[it]  = id >> 4;               // 0..31
        bnc[it]  = (id & 15) * 8;         // 0..120
    }
    v8h a_ld[2], b_ld[2];

    auto g_load = [&](int k0) {
#pragma unroll
        for (int it = 0; it < 2; ++it) {
            int ar = bm + arow[it]; if (ar >= M) ar = M - 1;
            a_ld[it] = ld8(A  + (size_t)ar * K + k0 + akc[it]);
            b_ld[it] = ld8(Bw + (size_t)(k0 + bkk[it]) * N + bn + bnc[it]);
        }
    };
    auto lds_store = [&]() {
#pragma unroll
        for (int it = 0; it < 2; ++it) {
            *(v8h*)(&As[arow[it]][akc[it]]) = a_ld[it];
#pragma unroll
            for (int e = 0; e < 8; ++e) Bs[bnc[it] + e][bkk[it]] = b_ld[it][e];
        }
    };

    v8f acc[2][4] = {};

    g_load(0);
    lds_store();
    for (int k0 = 0; k0 < K; k0 += TK) {
        __syncthreads();                       // tile ready in LDS
        // A fragments: lane row = lrow, chunks [kb,kb+8) and [kb+16,kb+24), kb = lhalf*8
        v16h af[2];
#pragma unroll
        for (int mi = 0; mi < 2; ++mi) {
            const f16_t* p = &As[wm + mi * 16 + lrow][lhalf * 8];
            af[mi] = concat8(ld8(p), ld8(p + 16));
        }
        // B fragments: lane col = lrow, chunks [kb,kb+8) and [kb+16,kb+24), kb = lhalf*8
        v16h bfr[4];
#pragma unroll
        for (int ni = 0; ni < 4; ++ni) {
            const f16_t* p = &Bs[wn + ni * 16 + lrow][lhalf * 8];
            bfr[ni] = concat8(ld8(p), ld8(p + 16));
        }
        const bool more = (k0 + TK) < K;
        if (more) {
            g_load(k0 + TK);                   // overlap global latency with WMMA
            if (k0 + 2 * TK < K) {             // prefetch tile after next
                int ar = bm + (tid >> 1); if (ar >= M) ar = M - 1;
                __builtin_prefetch(A  + (size_t)ar * K + (k0 + 2 * TK) + (tid & 1) * 16, 0, 1);
                __builtin_prefetch(Bw + (size_t)(k0 + 2 * TK + (tid >> 3)) * N + bn + (tid & 7) * 16, 0, 1);
            }
        }
#pragma unroll
        for (int mi = 0; mi < 2; ++mi)
#pragma unroll
            for (int ni = 0; ni < 4; ++ni)
                acc[mi][ni] = __builtin_amdgcn_wmma_f32_16x16x32_f16(
                    false, af[mi], false, bfr[ni], (short)0, acc[mi][ni], false, false);
        __syncthreads();                       // everyone done reading tile
        if (more) lds_store();
    }

    // epilogue: C layout -> lane<16: m=r, n=lane; lane>=16: m=8+r, n=lane-16
#pragma unroll
    for (int mi = 0; mi < 2; ++mi) {
#pragma unroll
        for (int ni = 0; ni < 4; ++ni) {
            int gn = bn + wn + ni * 16 + lrow;
#pragma unroll
            for (int r = 0; r < 8; ++r) {
                int gm = bm + wm + mi * 16 + lhalf * 8 + r;
                if (gm < M) {
                    float v = acc[mi][ni][r];
                    if (EPI == 1 || EPI == 3) v += bias[gn] + resid[(size_t)gm * N + gn];
                    if (EPI == 2) {
                        v += bias[gn];
                        v = 0.5f * v * (1.f + erff(v * 0.70710678118654752f));
                    }
                    if (EPI == 0 || EPI == 2) outB[(size_t)gm * N + gn] = (f16_t)v;
                    else                      outF[(size_t)gm * N + gn] = v;
                }
            }
        }
    }
}

// ---------------------------------------------------------------- flash attention
// one wave per (b, h, 16-query block); 32-key blocks with online softmax.
__global__ __launch_bounds__(128) void flash_attn(
    const f16_t* __restrict__ qkv, f16_t* __restrict__ o) {
    __shared__ __attribute__((aligned(16))) f16_t Pb[4][16][LDR];
    __shared__ __attribute__((aligned(16))) f16_t Vt[4][64][LDR];
    const int wave = threadIdx.x >> 5, lane = threadIdx.x & 31;
    const int task = blockIdx.x * 4 + wave;
    const int nqb = (NTOK + 15) / 16;            // 37
    const int qb = task % nqb;
    const int bh = task / nqb;
    const int h = bh % HEADS;
    const int batch = bh / HEADS;
    const int lrow = lane & 15, lhalf = lane >> 4;
    const float scale = 0.125f;                  // 1/sqrt(64)

    // Q fragments (16 rows x 64 d, 2 frags of 16x32)
    int qrow = qb * 16 + lrow;
    int qc = qrow < NTOK ? qrow : NTOK - 1;
    const f16_t* qptr = qkv + ((size_t)(batch * NTOK + qc)) * QKV_W + h * HEAD_DIM;
    v16h qa[2];
#pragma unroll
    for (int j = 0; j < 2; ++j) {
        int kb = j * 32 + lhalf * 8;
        qa[j] = concat8(ld8(qptr + kb), ld8(qptr + kb + 16));
    }

    v8f ot[4] = {};
    float mrow[8], lsum[8];
#pragma unroll
    for (int r = 0; r < 8; ++r) { mrow[r] = -3.0e38f; lsum[r] = 0.f; }

    f16_t (*P)[LDR] = Pb[wave];
    f16_t (*V)[LDR] = Vt[wave];

    for (int kb0 = 0; kb0 < NTOK; kb0 += 32) {
        // hoist all K fragments (8 x global_load_b128, one wait)
        v16h kf[2][2];
#pragma unroll
        for (int s = 0; s < 2; ++s) {
            int key = kb0 + s * 16 + lrow;
            int kc = key < NTOK ? key : NTOK - 1;
            const f16_t* kptr = qkv + ((size_t)(batch * NTOK + kc)) * QKV_W + DIM + h * HEAD_DIM;
#pragma unroll
            for (int j = 0; j < 2; ++j) {
                int db = j * 32 + lhalf * 8;
                kf[s][j] = concat8(ld8(kptr + db), ld8(kptr + db + 16));
            }
        }
        // S = scale * Q K^T : two 16x16 tiles
        v8f sc[2];
#pragma unroll
        for (int s = 0; s < 2; ++s) {
            v8f c = {};
#pragma unroll
            for (int j = 0; j < 2; ++j)
                c = __builtin_amdgcn_wmma_f32_16x16x32_f16(
                        false, qa[j], false, kf[s][j], (short)0, c, false, false);
            bool oob = (kb0 + s * 16 + lrow) >= NTOK;
#pragma unroll
            for (int r = 0; r < 8; ++r) c[r] = oob ? -3.0e38f : c[r] * scale;
            sc[s] = c;
        }
        // online softmax per row
        float p0[8], p1[8], corr[8];
#pragma unroll
        for (int r = 0; r < 8; ++r) {
            float rm = fmaxf(sc[0][r], sc[1][r]);
#pragma unroll
            for (int m = 8; m >= 1; m >>= 1) rm = fmaxf(rm, __shfl_xor(rm, m, 16));
            float nm = fmaxf(mrow[r], rm);
            corr[r] = __expf(mrow[r] - nm);
            p0[r] = __expf(sc[0][r] - nm);
            p1[r] = __expf(sc[1][r] - nm);
            float ps = p0[r] + p1[r];
#pragma unroll
            for (int m = 8; m >= 1; m >>= 1) ps += __shfl_xor(ps, m, 16);
            lsum[r] = lsum[r] * corr[r] + ps;
            mrow[r] = nm;
        }
#pragma unroll
        for (int t = 0; t < 4; ++t)
#pragma unroll
            for (int r = 0; r < 8; ++r) ot[t][r] *= corr[r];

        // P (C layout) -> LDS (row major 16x32)
#pragma unroll
        for (int r = 0; r < 8; ++r) {
            int m = lhalf * 8 + r;
            P[m][lrow]      = (f16_t)p0[r];
            P[m][16 + lrow] = (f16_t)p1[r];
        }
        // stage V block transposed: Vt[d][key_local], 64 x 32
#pragma unroll
        for (int u = 0; u < 8; ++u) {
            int id   = lane + 32 * u;      // 0..255
            int keyl = id & 31;
            int dch  = id >> 5;            // 0..7
            int key  = kb0 + keyl;
            if (key < NTOK) {
                const f16_t* vptr = qkv + ((size_t)(batch * NTOK + key)) * QKV_W
                                     + 2 * DIM + h * HEAD_DIM + dch * 8;
                v8h v = ld8(vptr);
#pragma unroll
                for (int e = 0; e < 8; ++e) V[dch * 8 + e][keyl] = v[e];
            } else {
#pragma unroll
                for (int e = 0; e < 8; ++e) V[dch * 8 + e][keyl] = (f16_t)0.f;
            }
        }
        // P fragment (A layout) + O += P V
        const f16_t* pp = &P[lrow][lhalf * 8];
        v16h pa = concat8(ld8(pp), ld8(pp + 16));
#pragma unroll
        for (int t = 0; t < 4; ++t) {
            const f16_t* vp = &V[t * 16 + lrow][lhalf * 8];
            v16h vb = concat8(ld8(vp), ld8(vp + 16));
            ot[t] = __builtin_amdgcn_wmma_f32_16x16x32_f16(
                        false, pa, false, vb, (short)0, ot[t], false, false);
        }
    }

    // O * (1/l) -> o_f16[token][h*64 + d]
    float inv[8];
#pragma unroll
    for (int r = 0; r < 8; ++r) inv[r] = 1.f / lsum[r];
#pragma unroll
    for (int t = 0; t < 4; ++t)
#pragma unroll
        for (int r = 0; r < 8; ++r) {
            int m = lhalf * 8 + r;
            int qr = qb * 16 + m;
            if (qr < NTOK) {
                float v = ot[t][r] * inv[r];
                o[((size_t)(batch * NTOK + qr)) * DIM + h * HEAD_DIM + t * 16 + lrow] = (f16_t)v;
            }
        }
}

// ---------------------------------------------------------------- launcher
extern "C" void kernel_launch(void* const* d_in, const int* in_sizes, int n_in,
                              void* d_out, int out_size, void* d_ws, size_t ws_size,
                              hipStream_t stream) {
    (void)in_sizes; (void)n_in; (void)out_size; (void)ws_size;
    const float* x     = (const float*)d_in[0];
    const float* ln1g  = (const float*)d_in[1];
    const float* ln1b  = (const float*)d_in[2];
    const float* Wqkv  = (const float*)d_in[3];
    const float* Wproj = (const float*)d_in[4];
    const float* bproj = (const float*)d_in[5];
    const float* ln2g  = (const float*)d_in[6];
    const float* ln2b  = (const float*)d_in[7];
    const float* W1    = (const float*)d_in[8];
    const float* b1    = (const float*)d_in[9];
    const float* W2    = (const float*)d_in[10];
    const float* b2    = (const float*)d_in[11];
    float* out = (float*)d_out;

    char* p = (char*)d_ws;
    auto alloc = [&](size_t bytes) -> void* {
        void* r = (void*)p;
        p += (bytes + 255) & ~(size_t)255;
        return r;
    };
    f16_t* wq  = (f16_t*)alloc((size_t)DIM * QKV_W  * 2);
    f16_t* wp  = (f16_t*)alloc((size_t)DIM * DIM    * 2);
    f16_t* w1  = (f16_t*)alloc((size_t)DIM * HIDDEN * 2);
    f16_t* w2  = (f16_t*)alloc((size_t)HIDDEN * DIM * 2);
    f16_t* h1  = (f16_t*)alloc((size_t)MROWS * DIM    * 2);
    f16_t* qkv = (f16_t*)alloc((size_t)MROWS * QKV_W  * 2);
    f16_t* ob  = (f16_t*)alloc((size_t)MROWS * DIM    * 2);
    float* x1  = (float*)alloc((size_t)MROWS * DIM    * 4);
    f16_t* h2  = (f16_t*)alloc((size_t)MROWS * DIM    * 2);
    f16_t* gb  = (f16_t*)alloc((size_t)MROWS * HIDDEN * 2);

    // weight conversion
    {
        int n;
        n = DIM * QKV_W;  cvt_f32_f16<<<(n + 255) / 256, 256, 0, stream>>>(Wqkv,  wq, n);
        n = DIM * DIM;    cvt_f32_f16<<<(n + 255) / 256, 256, 0, stream>>>(Wproj, wp, n);
        n = DIM * HIDDEN; cvt_f32_f16<<<(n + 255) / 256, 256, 0, stream>>>(W1,    w1, n);
        n = HIDDEN * DIM; cvt_f32_f16<<<(n + 255) / 256, 256, 0, stream>>>(W2,    w2, n);
    }
    const int lnBlocks = (MROWS + 7) / 8;
    const dim3 gemmGridQKV(QKV_W / TN,  (MROWS + TM - 1) / TM);
    const dim3 gemmGrid768(DIM / TN,    (MROWS + TM - 1) / TM);
    const dim3 gemmGridFC1(HIDDEN / TN, (MROWS + TM - 1) / TM);

    // LN1 -> h1 (f16)
    layernorm_to_f16<<<lnBlocks, 256, 0, stream>>>(x, ln1g, ln1b, h1, MROWS);
    // QKV gemm -> qkv (f16)
    gemm_f16_wmma<0><<<gemmGridQKV, 256, 0, stream>>>(
        h1, wq, nullptr, nullptr, nullptr, qkv, MROWS, QKV_W, DIM);
    // attention -> ob (f16)
    {
        int tasks = BATCH * HEADS * ((NTOK + 15) / 16);   // 28416
        flash_attn<<<tasks / 4, 128, 0, stream>>>(qkv, ob);
    }
    // proj + bias + residual(x) -> x1 (f32)
    gemm_f16_wmma<1><<<gemmGrid768, 256, 0, stream>>>(
        ob, wp, bproj, x, x1, nullptr, MROWS, DIM, DIM);
    // LN2 -> h2 (f16)
    layernorm_to_f16<<<lnBlocks, 256, 0, stream>>>(x1, ln2g, ln2b, h2, MROWS);
    // FC1 + bias + gelu -> gb (f16)
    gemm_f16_wmma<2><<<gemmGridFC1, 256, 0, stream>>>(
        h2, w1, b1, nullptr, nullptr, gb, MROWS, HIDDEN, DIM);
    // FC2 + bias + residual(x1) -> out (f32)
    gemm_f16_wmma<3><<<gemmGrid768, 256, 0, stream>>>(
        gb, w2, b2, x1, out, nullptr, MROWS, DIM, HIDDEN);
}
